// GraphAttentionLayer_16054587752786
// MI455X (gfx1250) — compile-verified, matrix operand order checked by reading, not run on hardware
//
#include <hip/hip_runtime.h>
#include <math.h>

typedef __attribute__((ext_vector_type(16))) __bf16 v16bf;
typedef __attribute__((ext_vector_type(8)))  float  v8f;
typedef __attribute__((ext_vector_type(4)))  unsigned int u32x4;
typedef __attribute__((ext_vector_type(8)))  int i32x8;
typedef __attribute__((ext_vector_type(4)))  int i32x4;

#define DMODEL 256
#define NNODES 400
#define NH 8
#define HDIM 32
#define NBATCH 32
#define NPAD 416  // 13 * 32: padded K for attn @ V, padded N stride for V^T

// K offset within a 16-bit 16x32 A/B fragment for VGPR-pair r, lane-half h.
// (CDNA5 ISA 7.12.2)
static __device__ __forceinline__ int koff(int r, int half) {
  return 2 * r + ((r >= 4) ? 8 : 0) + 8 * half;
}

// 16x32 bf16 fragment from a row-major bf16 row (unit K stride)
static __device__ __forceinline__ v16bf frag_from_bf16_row(const __bf16* rowp, int half) {
  v16bf f;
#pragma unroll
  for (int r = 0; r < 8; ++r) {
    int kb = koff(r, half);
    f[2 * r]     = rowp[kb];
    f[2 * r + 1] = rowp[kb + 1];
  }
  return f;
}

// 32x16 B fragment gathered from row-major f32 (K x N): one column per lane
static __device__ __forceinline__ v16bf frag_from_f32_col(const float* colp, int ldn, int half) {
  v16bf f;
#pragma unroll
  for (int r = 0; r < 8; ++r) {
    int kb = koff(r, half);
    f[2 * r]     = (__bf16)colp[kb * ldn];
    f[2 * r + 1] = (__bf16)colp[(kb + 1) * ldn];
  }
  return f;
}

static __device__ __forceinline__ v8f wmma_bf16(v16bf a, v16bf b, v8f c) {
  return __builtin_amdgcn_wmma_f32_16x16x32_bf16(false, a, false, b, (short)0, c,
                                                 false, false);
}

static __device__ __forceinline__ v8f vzero8() {
  v8f z = {0.f, 0.f, 0.f, 0.f, 0.f, 0.f, 0.f, 0.f};
  return z;
}

// ---------------------------------------------------------------------------
// Pack one weight matrix (f32 KxN row-major) into bf16 WMMA-B fragment order:
// wp[((kt*16 + nt)*32 + lane)*16 + e]  -> per-lane 32B contiguous (2x b128 load)
// ---------------------------------------------------------------------------
__global__ __launch_bounds__(32) void pack_w_kernel(const float* __restrict__ W,
                                                    __bf16* __restrict__ wp) {
  const int kt = blockIdx.x;           // 0..7   (K tiles of 32)
  const int nt = blockIdx.y;           // 0..15  (N tiles of 16)
  const int lane = threadIdx.x & 31;
  const int half = lane >> 4, l16 = lane & 15;
  v16bf f = frag_from_f32_col(W + (size_t)(kt * 32) * DMODEL + nt * 16 + l16,
                              DMODEL, half);
  *reinterpret_cast<v16bf*>(wp + (((size_t)kt * 16 + nt) * 32 + lane) * 16) = f;
}

// Elementwise x: f32 -> bf16 (vectorized float4 in, 4x bf16 out)
__global__ __launch_bounds__(256) void pack_x_kernel(const float* __restrict__ x,
                                                     __bf16* __restrict__ xb, int n) {
  int i = (blockIdx.x * 256 + threadIdx.x) * 4;
  if (i < n) {
    float4 t = *reinterpret_cast<const float4*>(x + i);
    xb[i]     = (__bf16)t.x;
    xb[i + 1] = (__bf16)t.y;
    xb[i + 2] = (__bf16)t.z;
    xb[i + 3] = (__bf16)t.w;
  }
}

// ---------------------------------------------------------------------------
// Projection GEMM: one wave computes a 32x64 strip of heads(x @ W + b).
// MODE 0: store bf16 in (b,h,n,d)    (for Q, K)
// MODE 1: store bf16 V transposed    (bh, d, n) with NPAD stride (tail pre-zeroed)
// ---------------------------------------------------------------------------
template <int MODE>
__global__ __launch_bounds__(32) void proj_kernel(
    const __bf16* __restrict__ xb, const __bf16* __restrict__ wp,
    const float* __restrict__ bias, __bf16* __restrict__ out) {
  const int lane = threadIdx.x & 31;
  const int half = lane >> 4, l16 = lane & 15;
  const int row0 = blockIdx.x * 32;
  const int ntb  = blockIdx.y * 4;     // first N-tile of this 64-col strip

  v8f acc[2][4];
#pragma unroll
  for (int s = 0; s < 2; ++s)
#pragma unroll
    for (int j = 0; j < 4; ++j) acc[s][j] = vzero8();

  const __bf16* xrow = xb + (size_t)(row0 + l16) * DMODEL;
  for (int kt = 0; kt < 8; ++kt) {
    v16bf a0 = frag_from_bf16_row(xrow + kt * 32, half);
    v16bf a1 = frag_from_bf16_row(xrow + 16 * DMODEL + kt * 32, half);
#pragma unroll
    for (int j = 0; j < 4; ++j) {
      v16bf b = *reinterpret_cast<const v16bf*>(
          wp + (((size_t)kt * 16 + ntb + j) * 32 + lane) * 16);
      acc[0][j] = wmma_bf16(a0, b, acc[0][j]);
      acc[1][j] = wmma_bf16(a1, b, acc[1][j]);
    }
  }

#pragma unroll
  for (int s = 0; s < 2; ++s) {
    const int r0   = row0 + s * 16;       // 16-row subtile: single batch (400%16==0)
    const int bidx = r0 / NNODES;
#pragma unroll
    for (int j = 0; j < 4; ++j) {
      int c  = (ntb + j) * 16 + l16;
      int hh = c >> 5, dd = c & 31;
      float bv = bias[c];
#pragma unroll
      for (int r = 0; r < 8; ++r) {
        int gr = r0 + r + 8 * half;
        int nn = gr - bidx * NNODES;
        float val = acc[s][j][r] + bv;
        if (MODE == 0)
          out[(((size_t)bidx * NH + hh) * NNODES + nn) * HDIM + dd] = (__bf16)val;
        else
          out[(((size_t)bidx * NH + hh) * HDIM + dd) * NPAD + nn] = (__bf16)val;
      }
    }
  }
}

// ---------------------------------------------------------------------------
// Attention for one (batch, head, 16-query strip). 4 waves.
// TDM: async 2D tile load of the (32 x 416) V^T slice into LDS at kernel
// entry (TENSORcnt), overlapped with QK^T WMMAs + softmax; drained before
// phase 3 consumes it.
// ---------------------------------------------------------------------------
__global__ __launch_bounds__(128) void attn_kernel(
    const __bf16* __restrict__ q, const __bf16* __restrict__ k,
    const __bf16* __restrict__ vt, const int* __restrict__ adj,
    __bf16* __restrict__ ctx) {
  __shared__ float  sS[16 * NNODES];
  __shared__ __bf16 sP[16 * NPAD];
  __shared__ __bf16 sV[HDIM * NPAD];     // V^T slice staged by TDM
  __shared__ float  sRed[16 * 8];
  __shared__ float  sMax[16];
  __shared__ float  sSum[16];

  const int bh   = blockIdx.x;           // 0..255
  const int bb   = bh >> 3, hh = bh & 7;
  const int qs   = blockIdx.y * 16;
  const int wave = threadIdx.x >> 5;
  const int lane = threadIdx.x & 31;
  const int half = lane >> 4;
  const int l16  = lane & 15;

  // ---- issue async TDM load: vt slice (32 rows x 416 bf16, contiguous) -> sV
  if (wave == 0) {
    unsigned lds_addr = (unsigned)(uintptr_t)(void*)sV;  // LDS aperture: low 32 bits
    unsigned long long ga =
        (unsigned long long)(uintptr_t)(vt + (size_t)bh * HDIM * NPAD);
    u32x4 g0;
    g0.x = 1u;                                          // count=1, user descriptor
    g0.y = lds_addr;                                    // lds_addr   [63:32]
    g0.z = (unsigned)(ga & 0xFFFFFFFFu);                // global_addr[31:0]
    g0.w = (unsigned)((ga >> 32) & 0x01FFFFFFu) | (2u << 30);  // addr[56:32], type=2
    i32x8 g1;
    g1[0] = 0x00010000;                  // wg_mask=0, data_size=1 (2 bytes)
    g1[1] = (int)(((unsigned)NPAD & 0xFFFFu) << 16);   // tensor_dim0[15:0] @ [63:48]
    g1[2] = (int)(((unsigned)HDIM & 0xFFFFu) << 16);   // dim0 hi=0, tensor_dim1[15:0]
    g1[3] = (int)(((unsigned)NPAD & 0xFFFFu) << 16);   // dim1 hi=0, tile_dim0=416
    g1[4] = (int)((unsigned)HDIM & 0xFFFFu);           // tile_dim1=32, tile_dim2=0
    g1[5] = (int)NPAD;                                 // tensor_dim0_stride lo32
    g1[6] = 0;                                         // stride0 hi, dim1_stride lo
    g1[7] = 0;
    i32x4 gz4 = {0, 0, 0, 0};
    i32x8 gz8 = {0, 0, 0, 0, 0, 0, 0, 0};
    // 6-arg form (clang-23 / therock-10.0 headers)
    __builtin_amdgcn_tensor_load_to_lds(g0, g1, gz4, gz4, gz8, 0);
  }

  const __bf16* qbh = q + (size_t)bh * NNODES * HDIM;
  const __bf16* kbh = k + (size_t)bh * NNODES * HDIM;
  const float scale = 0.17677669529663687f;  // 1/sqrt(HD)

  // warm the adjacency strip for the masking phase (global_prefetch_b8)
  {
    const char* adjs = (const char*)(adj + ((size_t)bb * NNODES + qs) * NNODES);
    for (int off = threadIdx.x * 128; off < 16 * NNODES * 4; off += 128 * 128)
      __builtin_prefetch(adjs + off, 0, 0);
  }

  v16bf aQ = frag_from_bf16_row(qbh + (size_t)(qs + l16) * HDIM, half);

  // Phase 1: score tiles (HD == K == 32 -> one WMMA per 16x16 tile)
  for (int t = wave; t < NNODES / 16; t += 4) {
    int m0 = t * 16;
    v16bf bK = frag_from_bf16_row(kbh + (size_t)(m0 + l16) * HDIM, half);
    v8f s = wmma_bf16(aQ, bK, vzero8());
    int key = m0 + l16;
#pragma unroll
    for (int r = 0; r < 8; ++r) {
      int lr = r + 8 * half;
      float val = s[r] * scale;
      if (adj[((size_t)bb * NNODES + (qs + lr)) * NNODES + key] == 0)
        val = -INFINITY;
      sS[lr * NNODES + key] = val;
    }
  }
  __syncthreads();

  // Phase 2: softmax. 8 threads per row, 50 elements each.
  {
    const int row = threadIdx.x >> 3;
    const int sub = threadIdx.x & 7;
    const int chunk = NNODES / 8;
    float mx = -INFINITY;
    for (int m = sub * chunk; m < (sub + 1) * chunk; ++m)
      mx = fmaxf(mx, sS[row * NNODES + m]);
    sRed[row * 8 + sub] = mx;
    __syncthreads();
    if (sub == 0) {
      float m2 = -INFINITY;
#pragma unroll
      for (int i = 0; i < 8; ++i) m2 = fmaxf(m2, sRed[row * 8 + i]);
      sMax[row] = m2;
    }
    __syncthreads();
    float rmax = sMax[row];
    float sum = 0.f;
    for (int m = sub * chunk; m < (sub + 1) * chunk; ++m) {
      float p = __expf(sS[row * NNODES + m] - rmax);
      __bf16 pb = (__bf16)p;
      sP[row * NPAD + m] = pb;
      sum += (float)pb;
    }
    if (sub == 0) {
#pragma unroll
      for (int m = NNODES; m < NPAD; ++m) sP[row * NPAD + m] = (__bf16)0.f;
    }
    sRed[row * 8 + sub] = sum;
    __syncthreads();
    if (sub == 0) {
      float s2 = 0.f;
#pragma unroll
      for (int i = 0; i < 8; ++i) s2 += sRed[row * 8 + i];
      sSum[row] = s2;
    }
  }

  // drain the TDM transfer (issuing wave), then make sV visible to all waves
  if (wave == 0) __builtin_amdgcn_s_wait_tensorcnt(0);
  __syncthreads();

  // Phase 3: ctx = P @ V from LDS (both operands LDS-resident)
  if (wave < 2) {
    const int d0 = wave * 16;
    const __bf16* vrow = &sV[(d0 + l16) * NPAD];
    v8f acc = vzero8();
    for (int kk = 0; kk < NPAD; kk += 32) {
      v16bf aP = frag_from_bf16_row(&sP[l16 * NPAD + kk], half);
      v16bf bV = frag_from_bf16_row(vrow + kk, half);
      acc = wmma_bf16(aP, bV, acc);
    }
#pragma unroll
    for (int r = 0; r < 8; ++r) {
      int lr = r + 8 * half;
      float o = acc[r] / sSum[lr];
      ctx[((size_t)bb * NNODES + (qs + lr)) * DMODEL + hh * HDIM + d0 + l16] =
          (__bf16)o;
    }
  }
}

// ---------------------------------------------------------------------------
// Output projection: 32x64 strip of ctx @ Wo + bo, f32 row-major output.
// ---------------------------------------------------------------------------
__global__ __launch_bounds__(32) void out_proj_kernel(
    const __bf16* __restrict__ ctx, const __bf16* __restrict__ wp,
    const float* __restrict__ bias, float* __restrict__ out) {
  const int lane = threadIdx.x & 31;
  const int half = lane >> 4, l16 = lane & 15;
  const int row0 = blockIdx.x * 32;
  const int ntb  = blockIdx.y * 4;

  v8f acc[2][4];
#pragma unroll
  for (int s = 0; s < 2; ++s)
#pragma unroll
    for (int j = 0; j < 4; ++j) acc[s][j] = vzero8();

  const __bf16* crow = ctx + (size_t)(row0 + l16) * DMODEL;
  for (int kt = 0; kt < 8; ++kt) {
    v16bf a0 = frag_from_bf16_row(crow + kt * 32, half);
    v16bf a1 = frag_from_bf16_row(crow + 16 * DMODEL + kt * 32, half);
#pragma unroll
    for (int j = 0; j < 4; ++j) {
      v16bf b = *reinterpret_cast<const v16bf*>(
          wp + (((size_t)kt * 16 + ntb + j) * 32 + lane) * 16);
      acc[0][j] = wmma_bf16(a0, b, acc[0][j]);
      acc[1][j] = wmma_bf16(a1, b, acc[1][j]);
    }
  }

#pragma unroll
  for (int s = 0; s < 2; ++s) {
#pragma unroll
    for (int j = 0; j < 4; ++j) {
      int c = (ntb + j) * 16 + l16;
      float bv = bias[c];
#pragma unroll
      for (int r = 0; r < 8; ++r) {
        int gr = row0 + s * 16 + r + 8 * half;
        out[(size_t)gr * DMODEL + c] = acc[s][j][r] + bv;
      }
    }
  }
}

extern "C" void kernel_launch(void* const* d_in, const int* in_sizes, int n_in,
                              void* d_out, int out_size, void* d_ws, size_t ws_size,
                              hipStream_t stream) {
  (void)in_sizes; (void)n_in; (void)out_size; (void)ws_size;
  const float* x  = (const float*)d_in[0];
  const int*   adj = (const int*)d_in[1];
  const float* Wq = (const float*)d_in[2];
  const float* bq = (const float*)d_in[3];
  const float* Wk = (const float*)d_in[4];
  const float* bk = (const float*)d_in[5];
  const float* Wv = (const float*)d_in[6];
  const float* bv = (const float*)d_in[7];
  const float* Wo = (const float*)d_in[8];
  const float* bo = (const float*)d_in[9];
  float* out = (float*)d_out;

  const size_t xelems = (size_t)NBATCH * NNODES * DMODEL;      // 3.28M
  const size_t welems = (size_t)DMODEL * DMODEL;               // 65536
  const size_t vtelems = (size_t)NBATCH * NH * HDIM * NPAD;    // 3.41M

  __bf16* xb  = (__bf16*)d_ws;
  __bf16* wqp = xb + xelems;
  __bf16* wkp = wqp + welems;
  __bf16* wvp = wkp + welems;
  __bf16* wop = wvp + welems;
  __bf16* qws = wop + welems;
  __bf16* kws = qws + xelems;
  __bf16* vtw = kws + xelems;
  __bf16* cws = vtw + vtelems;   // total ~33.6 MB

  pack_x_kernel<<<(int)(xelems / 1024), 256, 0, stream>>>(x, xb, (int)xelems);
  dim3 gw(8, 16);
  pack_w_kernel<<<gw, 32, 0, stream>>>(Wq, wqp);
  pack_w_kernel<<<gw, 32, 0, stream>>>(Wk, wkp);
  pack_w_kernel<<<gw, 32, 0, stream>>>(Wv, wvp);
  pack_w_kernel<<<gw, 32, 0, stream>>>(Wo, wop);

  (void)hipMemsetAsync(vtw, 0, vtelems * sizeof(__bf16), stream);  // zero V^T pad

  dim3 gp(NBATCH * NNODES / 32, DMODEL / 64);   // 400 x 4, 1 wave each
  proj_kernel<0><<<gp, 32, 0, stream>>>(xb, wqp, bq, qws);
  proj_kernel<0><<<gp, 32, 0, stream>>>(xb, wkp, bk, kws);
  proj_kernel<1><<<gp, 32, 0, stream>>>(xb, wvp, bv, vtw);

  dim3 ga(NBATCH * NH, NNODES / 16);            // 256 x 25 blocks, 4 waves each
  attn_kernel<<<ga, 128, 0, stream>>>(qws, kws, vtw, adj, cws);

  out_proj_kernel<<<gp, 32, 0, stream>>>(cws, wop, bo, out);
}
